// HybridGATLSTM_16664473109187
// MI455X (gfx1250) — compile-verified, matrix operand order checked by reading, not run on hardware
//
#include <hip/hip_runtime.h>
#include <hip/hip_bf16.h>
#include <math.h>

// ---------------------------------------------------------------------------
// HybridGATLSTM for MI455X (gfx1250, wave32, WMMA)
// B=4, S=512, N=128, HID=64, GATE=4*HID=256
// ---------------------------------------------------------------------------

typedef __attribute__((ext_vector_type(16))) _Float16 v16h;
typedef __attribute__((ext_vector_type(8)))  float    v8f;

#define BSZ   4
#define SSZ   512
#define NN    128
#define HIDC  64
#define GATEC 256
#define ROWS  (SSZ * BSZ)   // 2048

__device__ __forceinline__ float sigf(float x) { return 1.0f / (1.0f + __expf(-x)); }

// k index inside a 16x32 f16 WMMA fragment for (vgpr v, lane-half kb, half hf)
__device__ __forceinline__ int frag_k(int kc, int v, int kb, int hf) {
  return kc * 32 + ((v >> 2) * 16) + kb + ((v & 3) * 2) + hf;
}

// ---------------------------------------------------------------------------
// Input-side LSTM GEMM:  G[r][c] = sum_k A(r,k) * Wih[c][k] + bih[c] + bhh[c]
//   MODE 0: A(r,k) = x[b*S*N + s*N + k], b = r&3, s = r>>2  (K=128)
//   MODE 1: A(r,k) = hs[r*K + k]                            (K=64)
// One wave owns a 16-row stripe: A fragments loaded once, 16 column tiles
// swept with v_wmma_f32_16x16x32_f16.
// ---------------------------------------------------------------------------
template <int K, int MODE>
__global__ __launch_bounds__(32)
void lstm_in_gemm(const float* __restrict__ A, const float* __restrict__ Wih,
                  const float* __restrict__ bih, const float* __restrict__ bhh,
                  float* __restrict__ G) {
  constexpr int NKC = K / 32;
  int rt   = blockIdx.x;          // row tile 0..127
  int lane = threadIdx.x;         // 0..31
  int half = lane >> 4;
  int mloc = lane & 15;
  int kb   = half * 8;
  int m    = rt * 16 + mloc;

  const float* arow;
  if (MODE == 0) {
    int b = m & 3, s = m >> 2;
    arow = A + ((size_t)b * SSZ + (size_t)s) * NN;
  } else {
    arow = A + (size_t)m * (size_t)K;
  }

  // Load A fragments once (reused across all 16 column tiles)
  v16h afr[NKC];
#pragma unroll
  for (int kc = 0; kc < NKC; ++kc) {
#pragma unroll
    for (int v = 0; v < 8; ++v) {
      afr[kc][2 * v]     = (_Float16)arow[frag_k(kc, v, kb, 0)];
      afr[kc][2 * v + 1] = (_Float16)arow[frag_k(kc, v, kb, 1)];
    }
  }

#pragma unroll 1
  for (int ct = 0; ct < GATEC / 16; ++ct) {
    int n = ct * 16 + mloc;
    const float* brow = Wih + (size_t)n * (size_t)K;   // B[k][n] = Wih[n][k]
    v8f acc = {};
#pragma unroll
    for (int kc = 0; kc < NKC; ++kc) {
      v16h bf;
#pragma unroll
      for (int v = 0; v < 8; ++v) {
        bf[2 * v]     = (_Float16)brow[frag_k(kc, v, kb, 0)];
        bf[2 * v + 1] = (_Float16)brow[frag_k(kc, v, kb, 1)];
      }
      acc = __builtin_amdgcn_wmma_f32_16x16x32_f16(
          false, afr[kc], false, bf, (short)0, acc, false, false);
    }
    float bias = bih[n] + bhh[n];
#pragma unroll
    for (int v = 0; v < 8; ++v) {
      int row = rt * 16 + v + half * 8;
      G[(size_t)row * GATEC + n] = acc[v] + bias;
    }
  }
}

// ---------------------------------------------------------------------------
// Serial LSTM recurrence, WMMA edition. One block of 256 threads (8 waves).
// Whh is pre-converted to f16 B-fragments held in REGISTERS (4 x v16h per
// wave: 2 column tiles x 2 k-chunks) -> zero per-step weight traffic.
// Per step: rebuild the (4->16 row padded) h A-fragment from LDS, do 4 WMMAs
// per wave, then the elementwise gate update.
// ---------------------------------------------------------------------------
__global__ __launch_bounds__(256)
void lstm_layer(const float* __restrict__ G, const float* __restrict__ Whh,
                float* __restrict__ hs_out, float* __restrict__ hlast) {
  __shared__ float hsh[BSZ][HIDC + 1];   // stride 65: conflict-free frag gather
  __shared__ float csh[BSZ][HIDC];
  __shared__ float gsh[BSZ][GATEC];

  int t    = threadIdx.x;                // 0..255
  int lane = t & 31;
  int wave = t >> 5;                     // 0..7 -> col tiles 2w, 2w+1
  int half = lane >> 4;
  int mloc = lane & 15;
  int kb   = half * 8;

  // Preload B fragments of Whh^T (gate-major rows of Whh are B columns)
  v16h bfr[2][2];
#pragma unroll
  for (int c = 0; c < 2; ++c) {
    int n = (2 * wave + c) * 16 + mloc;          // gate column 0..255
    const float* brow = Whh + (size_t)n * HIDC;  // Whh[n][k]
#pragma unroll
    for (int kc = 0; kc < 2; ++kc) {
#pragma unroll
      for (int v = 0; v < 8; ++v) {
        bfr[c][kc][2 * v]     = (_Float16)brow[frag_k(kc, v, kb, 0)];
        bfr[c][kc][2 * v + 1] = (_Float16)brow[frag_k(kc, v, kb, 1)];
      }
    }
  }

  hsh[t >> 6][t & 63] = 0.0f;
  csh[t >> 6][t & 63] = 0.0f;
  __syncthreads();

  for (int s = 0; s < SSZ; ++s) {
    // ---- A fragment: rows = batch (0..3 valid, 4..15 zero), K = 64
    int mrow = mloc & 3;
    bool mval = (mloc < 4);
    v16h afr[2];
#pragma unroll
    for (int kc = 0; kc < 2; ++kc) {
#pragma unroll
      for (int v = 0; v < 8; ++v) {
        float h0 = hsh[mrow][frag_k(kc, v, kb, 0)];
        float h1 = hsh[mrow][frag_k(kc, v, kb, 1)];
        afr[kc][2 * v]     = (_Float16)(mval ? h0 : 0.0f);
        afr[kc][2 * v + 1] = (_Float16)(mval ? h1 : 0.0f);
      }
    }

    // ---- recurrent matmul: h @ Whh^T for this wave's 2 column tiles
#pragma unroll
    for (int c = 0; c < 2; ++c) {
      v8f acc = {};
      acc = __builtin_amdgcn_wmma_f32_16x16x32_f16(
          false, afr[0], false, bfr[c][0], (short)0, acc, false, false);
      acc = __builtin_amdgcn_wmma_f32_16x16x32_f16(
          false, afr[1], false, bfr[c][1], (short)0, acc, false, false);
      // D layout: lane(0..15)=col, VGPR v = row; rows 0..3 are the batches
      if (half == 0) {
        int col = (2 * wave + c) * 16 + mloc;
#pragma unroll
        for (int v = 0; v < 4; ++v) gsh[v][col] = acc[v];
      }
    }
    __syncthreads();

    // ---- elementwise gate update: thread t owns (b = t>>6, j = t&63)
    int b = t >> 6, j = t & 63;
    const float* gb = G + ((size_t)(s * 4 + b)) * GATEC;
    float ig = gsh[b][j]       + gb[j];
    float fg = gsh[b][64 + j]  + gb[64 + j];
    float gg = gsh[b][128 + j] + gb[128 + j];
    float og = gsh[b][192 + j] + gb[192 + j];
    float cn = sigf(fg) * csh[b][j] + sigf(ig) * tanhf(gg);
    float hn = sigf(og) * tanhf(cn);
    csh[b][j] = cn;
    hsh[b][j] = hn;
    if (hs_out) hs_out[((size_t)(s * 4 + b)) * HIDC + j] = hn;
    __syncthreads();
  }

  if (hlast) {
    int b = t >> 6, j = t & 63;
    hlast[b * HIDC + j] = hsh[b][j];
  }
}

// ---------------------------------------------------------------------------
// GAT branch: only the 4 graphs at s = S-1 are consumed by the heads.
// One block (128 threads = one per node) per graph.
// ---------------------------------------------------------------------------
__global__ __launch_bounds__(128)
void gat_kernel(const float* __restrict__ x, const int* __restrict__ adj,
                const float* __restrict__ emb_w, const float* __restrict__ emb_b,
                const float* __restrict__ W1, const float* __restrict__ a1,
                const float* __restrict__ W2, const float* __restrict__ a2,
                float* __restrict__ xg_out) {
  __shared__ float Wh1[NN][17];
  __shared__ float f1s[NN], f2s[NN];
  __shared__ float h1[NN][17];
  __shared__ float Wh2[NN][9];
  __shared__ float g1s[NN], g2s[NN];

  int b = blockIdx.x;     // graph (batch)
  int i = threadIdx.x;    // node

  float xv = x[((size_t)b * SSZ + (SSZ - 1)) * NN + i];
  float xe[16];
#pragma unroll
  for (int f = 0; f < 16; ++f) xe[f] = xv * emb_w[f] + emb_b[f];

  float wh[16];
#pragma unroll
  for (int f = 0; f < 16; ++f) {
    float s = 0.f;
#pragma unroll
    for (int k = 0; k < 16; ++k) s = fmaf(xe[k], W1[k * 16 + f], s);
    wh[f] = s;
    Wh1[i][f] = s;
  }
  float s1 = 0.f, s2 = 0.f;
#pragma unroll
  for (int f = 0; f < 16; ++f) { s1 = fmaf(wh[f], a1[f], s1); s2 = fmaf(wh[f], a1[16 + f], s2); }
  f1s[i] = s1; f2s[i] = s2;

  unsigned msk[4] = {0u, 0u, 0u, 0u};
  for (int j = 0; j < NN; ++j)
    if (adj[i * NN + j] > 0) msk[j >> 5] |= (1u << (j & 31));
  __syncthreads();

  // ---- GAT layer 1: masked softmax over j, aggregate Wh1, then ELU
  float base = f1s[i];
  float emax = -3.4e38f;
  for (int j = 0; j < NN; ++j) {
    if (msk[j >> 5] & (1u << (j & 31))) {
      float e = base + f2s[j];
      e = (e > 0.f) ? e : 0.2f * e;
      emax = fmaxf(emax, e);
    }
  }
  float ssum = 0.f;
  float acc[16];
#pragma unroll
  for (int f = 0; f < 16; ++f) acc[f] = 0.f;
  for (int j = 0; j < NN; ++j) {
    if (msk[j >> 5] & (1u << (j & 31))) {
      float e = base + f2s[j];
      e = (e > 0.f) ? e : 0.2f * e;
      float p = __expf(e - emax);
      ssum += p;
#pragma unroll
      for (int f = 0; f < 16; ++f) acc[f] = fmaf(p, Wh1[j][f], acc[f]);
    }
  }
  float inv = 1.0f / ssum;
#pragma unroll
  for (int f = 0; f < 16; ++f) {
    float v = acc[f] * inv;
    v = (v > 0.f) ? v : (__expf(v) - 1.0f);   // ELU (alpha=1)
    h1[i][f] = v;
  }
  __syncthreads();

  // ---- GAT layer 2 (F=8, no output activation)
  float wh2[8];
#pragma unroll
  for (int f = 0; f < 8; ++f) {
    float s = 0.f;
#pragma unroll
    for (int k = 0; k < 16; ++k) s = fmaf(h1[i][k], W2[k * 8 + f], s);
    wh2[f] = s;
    Wh2[i][f] = s;
  }
  float t1 = 0.f, t2 = 0.f;
#pragma unroll
  for (int f = 0; f < 8; ++f) { t1 = fmaf(wh2[f], a2[f], t1); t2 = fmaf(wh2[f], a2[8 + f], t2); }
  g1s[i] = t1; g2s[i] = t2;
  __syncthreads();

  base = g1s[i];
  emax = -3.4e38f;
  for (int j = 0; j < NN; ++j) {
    if (msk[j >> 5] & (1u << (j & 31))) {
      float e = base + g2s[j];
      e = (e > 0.f) ? e : 0.2f * e;
      emax = fmaxf(emax, e);
    }
  }
  ssum = 0.f;
  float acc8[8];
#pragma unroll
  for (int f = 0; f < 8; ++f) acc8[f] = 0.f;
  for (int j = 0; j < NN; ++j) {
    if (msk[j >> 5] & (1u << (j & 31))) {
      float e = base + g2s[j];
      e = (e > 0.f) ? e : 0.2f * e;
      float p = __expf(e - emax);
      ssum += p;
#pragma unroll
      for (int f = 0; f < 8; ++f) acc8[f] = fmaf(p, Wh2[j][f], acc8[f]);
    }
  }
  inv = 1.0f / ssum;
#pragma unroll
  for (int f = 0; f < 8; ++f)
    xg_out[b * (NN * 8) + i * 8 + f] = acc8[f] * inv;
}

// ---------------------------------------------------------------------------
// Heads: c = [xl (4x64) | xg (4x1024)] -> relu(c@w1+b1)@w2+b2 for 3 heads.
// d_out = dir(4) | ret(4) | vol(8), flat.
// ---------------------------------------------------------------------------
__global__ __launch_bounds__(128)
void heads_kernel(const float* __restrict__ xl, const float* __restrict__ xg,
                  const float* __restrict__ dw1, const float* __restrict__ db1,
                  const float* __restrict__ dw2, const float* __restrict__ db2,
                  const float* __restrict__ rw1, const float* __restrict__ rb1,
                  const float* __restrict__ rw2, const float* __restrict__ rb2,
                  const float* __restrict__ vw1, const float* __restrict__ vb1,
                  const float* __restrict__ vw2, const float* __restrict__ vb2,
                  float* __restrict__ out) {
  const int COMB = HIDC + NN * 8;          // 1088
  __shared__ float c[BSZ][COMB];
  __shared__ float hid[3][BSZ][32];
  int t = threadIdx.x;

  for (int idx = t; idx < BSZ * COMB; idx += 128) {
    int b = idx / COMB, k = idx % COMB;
    c[b][k] = (k < HIDC) ? xl[b * HIDC + k] : xg[b * (NN * 8) + (k - HIDC)];
  }
  __syncthreads();

  const float* w1s[3] = {dw1, rw1, vw1};
  const float* b1s[3] = {db1, rb1, vb1};
  for (int idx = t; idx < 3 * BSZ * 32; idx += 128) {
    int head = idx >> 7;
    int b    = (idx >> 5) & 3;
    int h    = idx & 31;
    const float* w1 = w1s[head];
    float s = b1s[head][h];
    for (int k = 0; k < COMB; ++k) s = fmaf(c[b][k], w1[k * 32 + h], s);
    hid[head][b][h] = fmaxf(s, 0.f);
  }
  __syncthreads();

  if (t < 16) {
    if (t < 4) {                        // dir: [4,1]
      float s = db2[0];
#pragma unroll
      for (int h = 0; h < 32; ++h) s = fmaf(hid[0][t][h], dw2[h], s);
      out[t] = s;
    } else if (t < 8) {                 // ret: [4,1]
      int b = t - 4;
      float s = rb2[0];
#pragma unroll
      for (int h = 0; h < 32; ++h) s = fmaf(hid[1][b][h], rw2[h], s);
      out[4 + b] = s;
    } else {                            // vol: [4,2]
      int b = (t - 8) >> 1, o = (t - 8) & 1;
      float s = vb2[o];
#pragma unroll
      for (int h = 0; h < 32; ++h) s = fmaf(hid[2][b][h], vw2[h * 2 + o], s);
      out[8 + b * 2 + o] = s;
    }
  }
}

// ---------------------------------------------------------------------------
extern "C" void kernel_launch(void* const* d_in, const int* in_sizes, int n_in,
                              void* d_out, int out_size, void* d_ws, size_t ws_size,
                              hipStream_t stream) {
  (void)in_sizes; (void)n_in; (void)out_size; (void)ws_size;

  const float* x      = (const float*)d_in[0];
  const int*   adj    = (const int*)  d_in[1];
  const float* emb_w  = (const float*)d_in[2];
  const float* emb_b  = (const float*)d_in[3];
  const float* gat1_W = (const float*)d_in[4];
  const float* gat1_a = (const float*)d_in[5];
  const float* gat2_W = (const float*)d_in[6];
  const float* gat2_a = (const float*)d_in[7];
  const float* Wih0   = (const float*)d_in[8];
  const float* Whh0   = (const float*)d_in[9];
  const float* bih0   = (const float*)d_in[10];
  const float* bhh0   = (const float*)d_in[11];
  const float* Wih1   = (const float*)d_in[12];
  const float* Whh1   = (const float*)d_in[13];
  const float* bih1   = (const float*)d_in[14];
  const float* bhh1   = (const float*)d_in[15];
  const float* dw1 = (const float*)d_in[16];
  const float* db1 = (const float*)d_in[17];
  const float* dw2 = (const float*)d_in[18];
  const float* db2 = (const float*)d_in[19];
  const float* rw1 = (const float*)d_in[20];
  const float* rb1 = (const float*)d_in[21];
  const float* rw2 = (const float*)d_in[22];
  const float* rb2 = (const float*)d_in[23];
  const float* vw1 = (const float*)d_in[24];
  const float* vb1 = (const float*)d_in[25];
  const float* vw2 = (const float*)d_in[26];
  const float* vb2 = (const float*)d_in[27];

  float* ws  = (float*)d_ws;
  float* G   = ws;                         // 2048*256
  float* HS0 = G + (size_t)ROWS * GATEC;   // 2048*64
  float* XG  = HS0 + (size_t)ROWS * HIDC;  // 4*1024
  float* XL  = XG + BSZ * NN * 8;          // 4*64

  const int rowTiles = ROWS / 16;          // 128 waves per GEMM

  // GAT branch (independent of LSTM, stream-ordered before heads)
  gat_kernel<<<BSZ, NN, 0, stream>>>(x, adj, emb_w, emb_b,
                                     gat1_W, gat1_a, gat2_W, gat2_a, XG);

  // LSTM layer 0: batched input GEMM (WMMA) + WMMA recurrence
  lstm_in_gemm<128, 0><<<rowTiles, 32, 0, stream>>>(x, Wih0, bih0, bhh0, G);
  lstm_layer<<<1, 256, 0, stream>>>(G, Whh0, HS0, nullptr);

  // LSTM layer 1: input GEMM over layer-0 hidden states (WMMA) + recurrence
  lstm_in_gemm<64, 1><<<rowTiles, 32, 0, stream>>>(HS0, Wih1, bih1, bhh1, G);
  lstm_layer<<<1, 256, 0, stream>>>(G, Whh1, nullptr, XL);

  // Heads -> d_out (16 floats: dir 4 | ret 4 | vol 8)
  heads_kernel<<<1, 128, 0, stream>>>(XL, XG,
                                      dw1, db1, dw2, db2,
                                      rw1, rb1, rw2, rb2,
                                      vw1, vb1, vw2, vb2,
                                      (float*)d_out);
}
